// Attention_84404697301229
// MI455X (gfx1250) — matrix operand order repaired and checked by reading, not compiled, not run on hardware
//
#include <hip/hip_runtime.h>

// ---------------------------------------------------------------------------
// Talking-heads attention, fused bf16-WMMA implementation for gfx1250.
//   b=16, n=1024, heads=12, dim_head=64, dim=768.
// Pipeline (all on `stream`):
//   1. convert x -> bf16, pack [Wq|Wkv] -> bf16 [768][2304], Wo -> bf16
//   2. QKV GEMM (wmma bf16): writes Q[b,h,n,d]*scale, K^T[b,h,d,n], V[b,h,n,d]
//   3. fused two-pass talking-heads attention (wave == head, LDS head mixing,
//      head-innermost LDS layout so cross-head mixes are vector ds_load_b128)
//   4. output GEMM (wmma bf16) + bias -> fp32 d_out
// ---------------------------------------------------------------------------

#define N_HEADS 12
#define D_HEAD 64
#define BATCH 16
#define SEQ 1024
#define DIM 768
#define QKV_N 2304
#define M_ROWS (BATCH * SEQ)   // 16384
#define SCALE_Q 0.125f         // 64^-0.5

typedef __attribute__((ext_vector_type(16))) __bf16 bf16x16;
typedef __attribute__((ext_vector_type(8)))  float  f32x8;

union BFrag { bf16x16 v; uint4 q[2]; };
union F8    { f32x8 v;   float f[8]; };
union F4V   { float4 v;  float f[4]; };
union BF8V  { uint4 q;   __bf16 e[8]; };
union BF4V  { uint2 q;   __bf16 e[4]; };

__device__ __forceinline__ f32x8 wmma_bf16(bf16x16 a, bf16x16 b, f32x8 c) {
  return __builtin_amdgcn_wmma_f32_16x16x32_bf16(false, a, false, b,
                                                 (short)0, c, false, false);
}

// A fragment (16x32 bf16) from row-major [M][K]; per ISA layout:
// lane<16: elems0-7 = K 0..7, elems8-15 = K 16..23 ; lane>=16: +8.
__device__ __forceinline__ bf16x16 load_a_rm(const __bf16* p, int ld,
                                             int row0, int col0, int lane) {
  BFrag f;
  const __bf16* base = p + (row0 + (lane & 15)) * ld + col0 + ((lane >> 4) << 3);
  f.q[0] = *(const uint4*)(base);
  f.q[1] = *(const uint4*)(base + 16);
  return f.v;
}

// B fragment (32x16 bf16) from row-major [K][N]; N = lane&15,
// lane<16: elems0-7 = K 0..7, elems8-15 = K 16..23 ; lane>=16: +8.
__device__ __forceinline__ bf16x16 load_b_km(const __bf16* p, int ld,
                                             int k0, int n0, int lane) {
  BFrag f;
  const __bf16* base = p + (k0 + ((lane >> 4) << 3)) * ld + n0 + (lane & 15);
#pragma unroll
  for (int s = 0; s < 8; ++s) {
    f.v[s] = base[s * ld];
    f.v[8 + s] = base[(16 + s) * ld];
  }
  return f.v;
}

__device__ __forceinline__ float hmax16(float v) {
#pragma unroll
  for (int m = 1; m < 16; m <<= 1) v = fmaxf(v, __shfl_xor(v, m, 32));
  return v;
}
__device__ __forceinline__ float hsum16(float v) {
#pragma unroll
  for (int m = 1; m < 16; m <<= 1) v += __shfl_xor(v, m, 32);
  return v;
}

// ------------------------------- converters --------------------------------
__global__ void cvt_bf16(const float* __restrict__ in, __bf16* __restrict__ o, int n) {
  int i = blockIdx.x * 256 + threadIdx.x;
  if (i < n) o[i] = (__bf16)in[i];
}

__global__ void pack_wqkv(const float* __restrict__ wq,
                          const float* __restrict__ wkv, __bf16* __restrict__ w) {
  int idx = blockIdx.x * 256 + threadIdx.x;
  if (idx >= DIM * QKV_N) return;
  int i = idx / QKV_N, j = idx % QKV_N;
  float v = (j < DIM) ? wq[i * DIM + j] : wkv[i * (2 * DIM) + (j - DIM)];
  w[idx] = (__bf16)v;
}

// ------------------------------- QKV GEMM ----------------------------------
// C[16384 x 2304] = xb @ W ; scatter into Q (scaled), K^T, V in bf16.
__global__ __launch_bounds__(256) void qkv_gemm(
    const __bf16* __restrict__ xb, const __bf16* __restrict__ w,
    __bf16* __restrict__ qb, __bf16* __restrict__ ktb, __bf16* __restrict__ vb) {
  const int lane = threadIdx.x & 31;
  const int wv = threadIdx.x >> 5;
  const int wm = wv >> 1, wn = wv & 1;          // 4 x 2 wave grid
  const int n0 = blockIdx.x * 64 + wn * 32;
  const int m0 = blockIdx.y * 128 + wm * 32;

  F8 acc[2][2];
#pragma unroll
  for (int a = 0; a < 2; ++a)
#pragma unroll
    for (int b = 0; b < 2; ++b)
#pragma unroll
      for (int r = 0; r < 8; ++r) acc[a][b].f[r] = 0.f;

  for (int kk = 0; kk < DIM; kk += 32) {
    bf16x16 a0 = load_a_rm(xb, DIM, m0, kk, lane);
    bf16x16 a1 = load_a_rm(xb, DIM, m0 + 16, kk, lane);
    bf16x16 b0 = load_b_km(w, QKV_N, kk, n0, lane);
    bf16x16 b1 = load_b_km(w, QKV_N, kk, n0 + 16, lane);
    acc[0][0].v = wmma_bf16(a0, b0, acc[0][0].v);
    acc[0][1].v = wmma_bf16(a0, b1, acc[0][1].v);
    acc[1][0].v = wmma_bf16(a1, b0, acc[1][0].v);
    acc[1][1].v = wmma_bf16(a1, b1, acc[1][1].v);
  }

  const int rhi = (lane >> 4) << 3, cn = lane & 15;
#pragma unroll
  for (int it = 0; it < 2; ++it)
#pragma unroll
    for (int nt = 0; nt < 2; ++nt)
#pragma unroll
      for (int r = 0; r < 8; ++r) {
        int m = m0 + it * 16 + r + rhi;
        int nc = n0 + nt * 16 + cn;
        int bb = m >> 10, nn = m & 1023;
        float v = acc[it][nt].f[r];
        if (nc < DIM) {                       // Q (fold softmax scale)
          int h = nc >> 6, dd = nc & 63;
          qb[(size_t)((bb * N_HEADS + h) * SEQ + nn) * D_HEAD + dd] =
              (__bf16)(v * SCALE_Q);
        } else if (nc < 2 * DIM) {            // K stored transposed [d][n]
          int j = nc - DIM;
          int h = j >> 6, dd = j & 63;
          ktb[(size_t)((bb * N_HEADS + h) * D_HEAD + dd) * SEQ + nn] = (__bf16)v;
        } else {                              // V natural [n][d]
          int j = nc - 2 * DIM;
          int h = j >> 6, dd = j & 63;
          vb[(size_t)((bb * N_HEADS + h) * SEQ + nn) * D_HEAD + dd] = (__bf16)v;
        }
      }
}

// --------------------------- fused attention -------------------------------
// One workgroup = one (batch, 32-row i tile); 12 waves = 12 heads.
// LDS (head-innermost so cross-head mixing is vectorized):
//   s_s : f32  scores [32*64 slots][20 floats] (12 used)  -> 163840 B
//   s_p : bf16 probs  [32*64 slots][24 bf16]  (12 used)   ->  98304 B
//   s_a : bf16 mixed-prob staging [12][32][72]            ->  55296 B
#define SS_STRIDE 20
#define SP_STRIDE 24
#define SA_STRIDE 72
#define SMEM_S (32 * 64 * SS_STRIDE * 4)
#define SMEM_P (32 * 64 * SP_STRIDE * 2)
#define SMEM_A (N_HEADS * 32 * SA_STRIDE * 2)
#define ATTN_SMEM (SMEM_S + SMEM_P + SMEM_A)   // 317440 <= 320KB

// compute raw scores for own head, publish to LDS, mix with mix_pre -> s[][]
__device__ __forceinline__ void scores_mix(
    const bf16x16 aq[2][2], const __bf16* kh, int j0, int lane, int g,
    float* s_s, const float* preg, F8 s[2][4]) {
#pragma unroll
  for (int it = 0; it < 2; ++it)
#pragma unroll
    for (int jt = 0; jt < 4; ++jt)
#pragma unroll
      for (int r = 0; r < 8; ++r) s[it][jt].f[r] = 0.f;

#pragma unroll
  for (int kc = 0; kc < 2; ++kc) {
#pragma unroll
    for (int jt = 0; jt < 4; ++jt) {
      bf16x16 bk = load_b_km(kh, SEQ, kc * 32, j0 + jt * 16, lane);
      s[0][jt].v = wmma_bf16(aq[0][kc], bk, s[0][jt].v);
      s[1][jt].v = wmma_bf16(aq[1][kc], bk, s[1][jt].v);
    }
  }
  const int rhi = (lane >> 4) << 3, cj = lane & 15;
#pragma unroll
  for (int it = 0; it < 2; ++it)
#pragma unroll
    for (int jt = 0; jt < 4; ++jt)
#pragma unroll
      for (int r = 0; r < 8; ++r) {
        int ii = it * 16 + r + rhi;
        int jj = jt * 16 + cj;
        s_s[(ii * 64 + jj) * SS_STRIDE + g] = s[it][jt].f[r];
      }
  __syncthreads();
#pragma unroll
  for (int it = 0; it < 2; ++it)
#pragma unroll
    for (int jt = 0; jt < 4; ++jt)
#pragma unroll
      for (int r = 0; r < 8; ++r) {
        int ii = it * 16 + r + rhi;
        int jj = jt * 16 + cj;
        const float* base = s_s + (ii * 64 + jj) * SS_STRIDE;
        F4V c0, c1, c2;
        c0.v = *(const float4*)(base);
        c1.v = *(const float4*)(base + 4);
        c2.v = *(const float4*)(base + 8);
        float a = 0.f;
#pragma unroll
        for (int h = 0; h < 4; ++h) a = fmaf(preg[h], c0.f[h], a);
#pragma unroll
        for (int h = 0; h < 4; ++h) a = fmaf(preg[4 + h], c1.f[h], a);
#pragma unroll
        for (int h = 0; h < 4; ++h) a = fmaf(preg[8 + h], c2.f[h], a);
        s[it][jt].f[r] = a;
      }
  __syncthreads();   // allow next chunk to overwrite s_s
}

__global__ __launch_bounds__(384) void attn_kernel(
    const __bf16* __restrict__ qb, const __bf16* __restrict__ ktb,
    const __bf16* __restrict__ vb, const float* __restrict__ mix_pre,
    const float* __restrict__ mix_post, __bf16* __restrict__ obuf) {
  extern __shared__ unsigned char smem[];
  float*  s_s = (float*)smem;
  __bf16* s_p = (__bf16*)(smem + SMEM_S);
  __bf16* s_a = (__bf16*)(smem + SMEM_S + SMEM_P);

  const int lane = threadIdx.x & 31;
  const int g    = threadIdx.x >> 5;   // head
  const int b    = blockIdx.y;
  const int i0   = blockIdx.x * 32;
  const int rhi  = (lane >> 4) << 3, cj = lane & 15;

  float preg[N_HEADS], postg[N_HEADS];
#pragma unroll
  for (int h = 0; h < N_HEADS; ++h) {
    preg[h]  = mix_pre[h * N_HEADS + g];
    postg[h] = mix_post[h * N_HEADS + g];
  }

  const __bf16* qh = qb  + (size_t)(b * N_HEADS + g) * SEQ * D_HEAD;
  const __bf16* kh = ktb + (size_t)(b * N_HEADS + g) * D_HEAD * SEQ;
  const __bf16* vh = vb  + (size_t)(b * N_HEADS + g) * SEQ * D_HEAD;

  bf16x16 aq[2][2];
#pragma unroll
  for (int it = 0; it < 2; ++it)
#pragma unroll
    for (int kc = 0; kc < 2; ++kc)
      aq[it][kc] = load_a_rm(qh, D_HEAD, i0 + it * 16, kc * 32, lane);

  float m[2][8], z[2][8];
#pragma unroll
  for (int it = 0; it < 2; ++it)
#pragma unroll
    for (int r = 0; r < 8; ++r) { m[it][r] = -1e30f; z[it][r] = 0.f; }

  F8 s[2][4];

  // ---- pass 1: exact row max + normalizer ----
#pragma unroll 1
  for (int jc = 0; jc < SEQ / 64; ++jc) {
    scores_mix(aq, kh, jc * 64, lane, g, s_s, preg, s);
#pragma unroll
    for (int it = 0; it < 2; ++it)
#pragma unroll
      for (int r = 0; r < 8; ++r) {
        float mx = s[it][0].f[r];
#pragma unroll
        for (int jt = 1; jt < 4; ++jt) mx = fmaxf(mx, s[it][jt].f[r]);
        mx = hmax16(mx);
        float mo = m[it][r];
        float mn = fmaxf(mo, mx);
        float rs = 0.f;
#pragma unroll
        for (int jt = 0; jt < 4; ++jt) rs += __expf(s[it][jt].f[r] - mn);
        rs = hsum16(rs);
        z[it][r] = z[it][r] * __expf(mo - mn) + rs;
        m[it][r] = mn;
      }
  }
#pragma unroll
  for (int it = 0; it < 2; ++it)
#pragma unroll
    for (int r = 0; r < 8; ++r) z[it][r] = 1.f / z[it][r];   // -> 1/Z

  // ---- pass 2: normalized P, post-mix, P@V ----
  F8 o[2][4];
#pragma unroll
  for (int it = 0; it < 2; ++it)
#pragma unroll
    for (int dt = 0; dt < 4; ++dt)
#pragma unroll
      for (int r = 0; r < 8; ++r) o[it][dt].f[r] = 0.f;

#pragma unroll 1
  for (int jc = 0; jc < SEQ / 64; ++jc) {
    const int j0 = jc * 64;
    scores_mix(aq, kh, j0, lane, g, s_s, preg, s);

    // exactly-normalized softmax probabilities for own head
#pragma unroll
    for (int it = 0; it < 2; ++it)
#pragma unroll
      for (int jt = 0; jt < 4; ++jt)
#pragma unroll
        for (int r = 0; r < 8; ++r) {
          int ii = it * 16 + r + rhi;
          int jj = jt * 16 + cj;
          float p = __expf(s[it][jt].f[r] - m[it][r]) * z[it][r];
          s_p[(ii * 64 + jj) * SP_STRIDE + g] = (__bf16)p;
        }
    __syncthreads();

    // post-softmax talking-heads mix: A_g = sum_h post[h,g] * P_h
#pragma unroll
    for (int it = 0; it < 2; ++it)
#pragma unroll
      for (int jt = 0; jt < 4; ++jt)
#pragma unroll
        for (int r = 0; r < 8; ++r) {
          int ii = it * 16 + r + rhi;
          int jj = jt * 16 + cj;
          const __bf16* base = s_p + (ii * 64 + jj) * SP_STRIDE;
          BF8V lo; lo.q = *(const uint4*)(base);
          BF4V hi; hi.q = *(const uint2*)(base + 8);
          float a = 0.f;
#pragma unroll
          for (int h = 0; h < 8; ++h) a = fmaf(postg[h], (float)lo.e[h], a);
#pragma unroll
          for (int h = 0; h < 4; ++h) a = fmaf(postg[8 + h], (float)hi.e[h], a);
          s_a[(g * 32 + ii) * SA_STRIDE + jj] = (__bf16)a;
        }
    __syncthreads();

    // O_g += A_g @ V_g
#pragma unroll
    for (int kc = 0; kc < 2; ++kc) {
      bf16x16 a0 = load_a_rm(s_a + g * 32 * SA_STRIDE, SA_STRIDE, 0,  kc * 32, lane);
      bf16x16 a1 = load_a_rm(s_a + g * 32 * SA_STRIDE, SA_STRIDE, 16, kc * 32, lane);
#pragma unroll
      for (int dt = 0; dt < 4; ++dt) {
        bf16x16 bv = load_b_km(vh, D_HEAD, j0 + kc * 32, dt * 16, lane);
        o[0][dt].v = wmma_bf16(a0, bv, o[0][dt].v);
        o[1][dt].v = wmma_bf16(a1, bv, o[1][dt].v);
      }
    }
    __syncthreads();
  }

  // write head-g slice of the concatenated activation (bf16, row-major [n][768])
#pragma unroll
  for (int it = 0; it < 2; ++it)
#pragma unroll
    for (int dt = 0; dt < 4; ++dt)
#pragma unroll
      for (int r = 0; r < 8; ++r) {
        int row = i0 + it * 16 + r + rhi;
        int col = g * D_HEAD + dt * 16 + cj;
        obuf[(size_t)(b * SEQ + row) * DIM + col] = (__bf16)o[it][dt].f[r];
      }
}

// ------------------------------ output GEMM --------------------------------
__global__ __launch_bounds__(256) void out_gemm(
    const __bf16* __restrict__ ob, const __bf16* __restrict__ wo,
    const float* __restrict__ bo, float* __restrict__ out) {
  const int lane = threadIdx.x & 31;
  const int wv = threadIdx.x >> 5;
  const int wm = wv >> 1, wn = wv & 1;
  const int n0 = blockIdx.x * 64 + wn * 32;
  const int m0 = blockIdx.y * 128 + wm * 32;

  F8 acc[2][2];
#pragma unroll
  for (int a = 0; a < 2; ++a)
#pragma unroll
    for (int b2 = 0; b2 < 2; ++b2)
#pragma unroll
      for (int r = 0; r < 8; ++r) acc[a][b2].f[r] = 0.f;

  for (int kk = 0; kk < DIM; kk += 32) {
    bf16x16 a0 = load_a_rm(ob, DIM, m0, kk, lane);
    bf16x16 a1 = load_a_rm(ob, DIM, m0 + 16, kk, lane);
    bf16x16 b0 = load_b_km(wo, DIM, kk, n0, lane);
    bf16x16 b1 = load_b_km(wo, DIM, kk, n0 + 16, lane);
    acc[0][0].v = wmma_bf16(a0, b0, acc[0][0].v);
    acc[0][1].v = wmma_bf16(a0, b1, acc[0][1].v);
    acc[1][0].v = wmma_bf16(a1, b0, acc[1][0].v);
    acc[1][1].v = wmma_bf16(a1, b1, acc[1][1].v);
  }

  const int rhi = (lane >> 4) << 3, cn = lane & 15;
#pragma unroll
  for (int it = 0; it < 2; ++it)
#pragma unroll
    for (int nt = 0; nt < 2; ++nt)
#pragma unroll
      for (int r = 0; r < 8; ++r) {
        int mrow = m0 + it * 16 + r + rhi;
        int nc = n0 + nt * 16 + cn;
        out[(size_t)mrow * DIM + nc] = acc[it][nt].f[r] + bo[nc];
      }
}

// -------------------------------- launcher ---------------------------------
extern "C" void kernel_launch(void* const* d_in, const int* in_sizes, int n_in,
                              void* d_out, int out_size, void* d_ws, size_t ws_size,
                              hipStream_t stream) {
  (void)in_sizes; (void)n_in; (void)out_size; (void)ws_size;
  const float* x        = (const float*)d_in[0];
  const float* Wq       = (const float*)d_in[1];
  const float* Wkv      = (const float*)d_in[2];
  const float* mix_pre  = (const float*)d_in[3];
  const float* mix_post = (const float*)d_in[4];
  const float* Wo       = (const float*)d_in[5];
  const float* bo       = (const float*)d_in[6];
  float* out = (float*)d_out;

  unsigned char* ws = (unsigned char*)d_ws;
  size_t off = 0;
  __bf16* xb    = (__bf16*)(ws + off); off += (size_t)M_ROWS * DIM * 2;
  __bf16* wqkvb = (__bf16*)(ws + off); off += (size_t)DIM * QKV_N * 2;
  __bf16* wob   = (__bf16*)(ws + off); off += (size_t)DIM * DIM * 2;
  __bf16* qb    = (__bf16*)(ws + off); off += (size_t)M_ROWS * DIM * 2;
  __bf16* ktb   = (__bf16*)(ws + off); off += (size_t)M_ROWS * DIM * 2;
  __bf16* vb    = (__bf16*)(ws + off); off += (size_t)M_ROWS * DIM * 2;
  __bf16* obuf  = (__bf16*)(ws + off); off += (size_t)M_ROWS * DIM * 2;

  cvt_bf16<<<(M_ROWS * DIM + 255) / 256, 256, 0, stream>>>(x, xb, M_ROWS * DIM);
  pack_wqkv<<<(DIM * QKV_N + 255) / 256, 256, 0, stream>>>(Wq, Wkv, wqkvb);
  cvt_bf16<<<(DIM * DIM + 255) / 256, 256, 0, stream>>>(Wo, wob, DIM * DIM);

  qkv_gemm<<<dim3(QKV_N / 64, M_ROWS / 128), 256, 0, stream>>>(xb, wqkvb, qb, ktb, vb);

  attn_kernel<<<dim3(SEQ / 32, BATCH), 384, ATTN_SMEM, stream>>>(
      qb, ktb, vb, mix_pre, mix_post, obuf);

  out_gemm<<<dim3(DIM / 64, M_ROWS / 128), 256, 0, stream>>>(obuf, wob, bo, out);
}
